// Model_24541443130121
// MI455X (gfx1250) — matrix operand order repaired and checked by reading, not compiled, not run on hardware
//
#include <hip/hip_runtime.h>
#include <hip/hip_bf16.h>
#include <math.h>

#define B_   256
#define T_   784
#define H_   512
#define NCLS_ 10
#define GRID_RNN 64          // persistent workgroups (must be co-resident)
#define WAVES_PER_BLK 8      // 256 threads / wave32
#define KBLK 16              // 512 / 32
#define AROW 520             // LDS row pitch in bf16: 512 + 8 (16B TDM pad) -> conflict-free-ish

typedef __attribute__((ext_vector_type(16))) __bf16 v16bf;
typedef __attribute__((ext_vector_type(8)))  __bf16 v8bf;
typedef __attribute__((ext_vector_type(8)))  float  v8f;
typedef __attribute__((ext_vector_type(4)))  unsigned int u32x4;
typedef __attribute__((ext_vector_type(8)))  int i32x8;
typedef __attribute__((ext_vector_type(4)))  int i32x4;

// ---------------------------------------------------------------------------
// Prep: W_hh f32 -> bf16, zero h0 (bf16), transpose inputs through `order`
// into xT[t*B + b], zero the grid-barrier counter.
// ---------------------------------------------------------------------------
__global__ void prep_kernel(const float* __restrict__ Whh,
                            const float* __restrict__ inputs,     // B x T
                            const long long* __restrict__ order,  // T
                            __bf16* __restrict__ Wbf,
                            __bf16* __restrict__ h0,
                            float*  __restrict__ xT,              // T x B
                            unsigned int* __restrict__ counter) {
    int idx = blockIdx.x * blockDim.x + threadIdx.x;
    if (idx < H_ * H_) Wbf[idx] = (__bf16)Whh[idx];
    if (idx < B_ * H_) h0[idx]  = (__bf16)0.0f;
    if (idx < T_ * B_) {
        int t = idx / B_;
        int m = idx - t * B_;
        xT[idx] = inputs[m * T_ + (int)order[t]];
    }
    if (idx == 0) *counter = 0u;
}

// branchless tanh: 1 - 2/(exp(2x)+1); saturates correctly at +/-1
__device__ __forceinline__ float fast_tanh(float x) {
    float e2 = __expf(x + x);
    return 1.0f - 2.0f * __builtin_amdgcn_rcpf(e2 + 1.0f);
}

// Build TDM descriptor group0 for a 16-row x 1024B tile at `ga`, LDS at lds_off.
__device__ __forceinline__ u32x4 make_g0(unsigned long long ga, unsigned lds_off) {
    u32x4 g0;
    g0.x = 1u;                                   // count=1, user descriptor
    g0.y = lds_off;                              // lds_addr
    g0.z = (unsigned)(ga & 0xffffffffu);         // global_addr[31:0]
    g0.w = (unsigned)(ga >> 32) | (2u << 30);    // global_addr[56:32] | type=2
    return g0;
}

// ---------------------------------------------------------------------------
// Persistent recurrent kernel.
//  - B slice of W_hh register-resident (128 VGPRs) across all 784 steps.
//  - A tile (h, 16 rows x 512 bf16 = 16KB, shared by all 8 waves of the WG)
//    staged into LDS once per step by ONE tensor_load_to_lds (TDM DMA) with
//    16B/row padding, then consumed via ds_load_b128.
//  - Grid-wide barrier between steps via monotonic atomic counter.
// ---------------------------------------------------------------------------
__global__ void __launch_bounds__(256, 1)
rnn_kernel(const float* __restrict__ xT,          // T x B f32
           const float* __restrict__ w_ih,        // H f32 (W_ih[:,0])
           const float* __restrict__ b_ih,        // H f32
           const float* __restrict__ b_hh,        // H f32
           const __bf16* __restrict__ Wbf,        // H x H bf16
           __bf16* __restrict__ hbuf0,            // B x H bf16
           __bf16* __restrict__ hbuf1,            // B x H bf16
           unsigned int* __restrict__ counter) {
    __shared__ __align__(16) __bf16 shA[16 * AROW];   // 16.25 KB padded A tile

    const int lane = threadIdx.x & 31;
    const int wave = threadIdx.x >> 5;
    const int gw   = blockIdx.x * WAVES_PER_BLK + wave;   // 0..511
    const int mt   = gw >> 5;                             // 0..15 row tile (uniform per block)
    const int nt   = gw & 31;                             // 0..31 col tile
    const int half = lane >> 4;
    const int l16  = lane & 15;

    const int n     = nt * 16 + l16;        // hidden unit (B/C/D column) of this lane
    const int mbase = mt * 16 + half * 8;   // first of 8 C/D rows this lane owns

    const float wih_n  = w_ih[n];
    const float bias_n = b_ih[n] + b_hh[n];

    // ---- register-cache the wave's B slice of W_hh for the whole scan ----
    v16bf Breg[KBLK];
    const __bf16* brow = Wbf + n * H_;
    #pragma unroll
    for (int kb = 0; kb < KBLK; ++kb)
        Breg[kb] = __builtin_shufflevector(*(const v8bf*)(brow + kb * 32 + half * 8),
                                           *(const v8bf*)(brow + kb * 32 + 16 + half * 8),
                                           0,1,2,3,4,5,6,7,8,9,10,11,12,13,14,15);

    // ---- TDM descriptors (ping/pong on h buffer), built once ----
    // group1: data_size=8B, pad_enable, pad_interval=256 DW (1024B = 1 row),
    // pad_amount=4 DW (16B); tensor/tile: dim0=128 units, dim1=16 rows.
    i32x8 g1;
    g1[0] = (3 << 16) | (1 << 20) | (7 << 22) | (3 << 25);
    g1[1] = (128 << 16);          // tensor_dim0[15:0] at bit 48
    g1[2] = (16 << 16);           // tensor_dim1 at bit 80
    g1[3] = (128 << 16);          // tile_dim0 at bit 112
    g1[4] = 16;                   // tile_dim1 at bit 128
    g1[5] = 128;                  // tensor_dim0_stride at bit 160
    g1[6] = (int)(2048u << 16);   // tensor_dim1_stride at bit 208
    g1[7] = 0;
    i32x4 gz4 = {};
    i32x8 gz8 = {};
    const unsigned lds_off = (unsigned)(unsigned long long)(void*)shA;
    const unsigned long long tile_bytes = (unsigned long long)(mt * 16) * H_ * 2ull;
    const u32x4 g0_even = make_g0((unsigned long long)hbuf0 + tile_bytes, lds_off);
    const u32x4 g0_odd  = make_g0((unsigned long long)hbuf1 + tile_bytes, lds_off);

    const __bf16* hc = hbuf0;   // h_t   (read)
    __bf16*       hn = hbuf1;   // h_t+1 (write)

    for (int t = 0; t < T_; ++t) {
        // ---- stage A tile (h) into LDS via Tensor Data Mover ----
        if (wave == 0) {
            __builtin_amdgcn_tensor_load_to_lds((t & 1) ? g0_odd : g0_even,
                                                g1, gz4, gz4, gz8, 0);
            __builtin_amdgcn_s_wait_tensorcnt(0);
        }
        __syncthreads();

        // ---- GEMM tile: c = h[mt-tile, :] @ W_hh.T[:, nt-tile], f32 accum ----
        v8f c = {};
        const __bf16* arow = shA + l16 * AROW;   // A row of this lane, in LDS
        #pragma unroll
        for (int kb = 0; kb < KBLK; ++kb) {
            const int k0 = kb * 32;
            v8bf alo = *(const v8bf*)(arow + k0 + half * 8);
            v8bf ahi = *(const v8bf*)(arow + k0 + 16 + half * 8);
            v16bf a = __builtin_shufflevector(alo, ahi,
                        0,1,2,3,4,5,6,7,8,9,10,11,12,13,14,15);
            c = __builtin_amdgcn_wmma_f32_16x16x32_bf16(
                    false, a, false, Breg[kb], (short)0, c, false, false);
        }

        // ---- epilogue: + x_t * w_ih + biases, tanh, store bf16 ----
        const float4* xp = (const float4*)(xT + (size_t)t * B_ + mbase);
        float4 x0 = xp[0];
        float4 x1 = xp[1];
        float xv[8] = { x0.x, x0.y, x0.z, x0.w, x1.x, x1.y, x1.z, x1.w };
        #pragma unroll
        for (int v = 0; v < 8; ++v) {
            const int m   = mbase + v;                     // C/D row: M = half*8 + v
            const float val = c[v] + xv[v] * wih_n + bias_n;
            hn[m * H_ + n] = (__bf16)fast_tanh(val);
        }

        // ---- grid-wide barrier (release -> signal -> spin -> acquire) ----
        __syncthreads();
        if (threadIdx.x == 0) {
            __threadfence();                 // make h_next visible device-wide
            atomicAdd(counter, 1u);
            const unsigned target = (unsigned)(t + 1) * (unsigned)GRID_RNN;
            while (__hip_atomic_load(counter, __ATOMIC_RELAXED,
                                     __HIP_MEMORY_SCOPE_AGENT) < target) {
                __builtin_amdgcn_s_sleep(1);
            }
            __threadfence();                 // invalidate stale near caches
        }
        __syncthreads();

        // swap buffers
        const __bf16* tmp = hc; hc = hn; hn = (__bf16*)tmp;
    }
    // T_ = 784 (even) -> final h resides in hbuf0
}

// ---------------------------------------------------------------------------
// Head: out = h @ lin_W.T + lin_b; log_softmax; loss = -mean(logp[b, y[b]]);
// correct = sum(argmax == y). One thread per batch row, LDS tree reduction.
// ---------------------------------------------------------------------------
__global__ void head_kernel(const __bf16* __restrict__ h,       // B x H bf16
                            const float* __restrict__ linW,     // NCLS x H
                            const float* __restrict__ linb,     // NCLS
                            const long long* __restrict__ y,    // B
                            float* __restrict__ out) {
    __shared__ float sloss[B_];
    __shared__ int   scorr[B_];
    const int b = threadIdx.x;

    float o[NCLS_];
    #pragma unroll
    for (int ci = 0; ci < NCLS_; ++ci) {
        float acc = linb[ci];
        const float* wr = linW + ci * H_;
        for (int k = 0; k < H_; ++k) acc += (float)h[b * H_ + k] * wr[k];
        o[ci] = acc;
    }
    float mx = o[0]; int pred = 0;
    #pragma unroll
    for (int ci = 1; ci < NCLS_; ++ci)
        if (o[ci] > mx) { mx = o[ci]; pred = ci; }   // first-max tiebreak like argmax
    float s = 0.0f;
    #pragma unroll
    for (int ci = 0; ci < NCLS_; ++ci) s += expf(o[ci] - mx);
    const float lse = mx + logf(s);
    const int yb = (int)y[b];
    sloss[b] = -(o[yb] - lse);
    scorr[b] = (pred == yb) ? 1 : 0;
    __syncthreads();
    for (int stride = B_ / 2; stride > 0; stride >>= 1) {
        if (b < stride) { sloss[b] += sloss[b + stride]; scorr[b] += scorr[b + stride]; }
        __syncthreads();
    }
    if (b == 0) {
        out[0] = sloss[0] / (float)B_;   // loss
        out[1] = (float)scorr[0];        // correct count
    }
}

// ---------------------------------------------------------------------------
extern "C" void kernel_launch(void* const* d_in, const int* in_sizes, int n_in,
                              void* d_out, int out_size, void* d_ws, size_t ws_size,
                              hipStream_t stream) {
    const float*     inputs = (const float*)    d_in[0];   // B x T
    const long long* y      = (const long long*)d_in[1];   // B
    const long long* order  = (const long long*)d_in[2];   // T
    const float*     W_ih   = (const float*)    d_in[3];   // H x 1
    const float*     b_ih   = (const float*)    d_in[4];   // H
    const float*     W_hh   = (const float*)    d_in[5];   // H x H
    const float*     b_hh   = (const float*)    d_in[6];   // H
    const float*     lin_W  = (const float*)    d_in[7];   // NCLS x H
    const float*     lin_b  = (const float*)    d_in[8];   // NCLS

    char* ws = (char*)d_ws;
    size_t off = 0;
    unsigned int* counter = (unsigned int*)(ws + off); off += 256;
    __bf16* Wbf = (__bf16*)(ws + off); off += (size_t)H_ * H_ * 2;   // 512 KB
    __bf16* h0  = (__bf16*)(ws + off); off += (size_t)B_ * H_ * 2;   // 256 KB
    __bf16* h1  = (__bf16*)(ws + off); off += (size_t)B_ * H_ * 2;   // 256 KB
    float*  xT  = (float*) (ws + off); off += (size_t)T_ * B_ * 4;   // 784 KB

    prep_kernel<<<(H_ * H_ + 255) / 256, 256, 0, stream>>>(W_hh, inputs, order,
                                                           Wbf, h0, xT, counter);
    rnn_kernel<<<GRID_RNN, 256, 0, stream>>>(xT, W_ih, b_ih, b_hh,
                                             Wbf, h0, h1, counter);
    head_kernel<<<1, B_, 0, stream>>>(h0, lin_W, lin_b, y, (float*)d_out);
}
